// RGAT_82360292868669
// MI455X (gfx1250) — matrix operand order repaired and checked by reading, not compiled, not run on hardware
//
#include <hip/hip_runtime.h>
#include <math.h>

#define NHEADS 4
#define HD 64
#define NEG_SLOPE 0.2f

typedef __attribute__((ext_vector_type(16))) __bf16 v16bf;
typedef __attribute__((ext_vector_type(8)))  __bf16 v8bf;
typedef __attribute__((ext_vector_type(8)))  float  v8f;

// ---------- helpers ----------
__device__ __forceinline__ void atomicMaxF(float* addr, float v) {
    // Standard monotone-bit-pattern trick: works with init value -inf.
    if (v >= 0.f) atomicMax((int*)addr, __float_as_int(v));
    else          atomicMin((unsigned int*)addr, __float_as_uint(v));
}

// ---------- conversion / packing ----------
__global__ void k_f32_to_bf16(const float* __restrict__ x, __bf16* __restrict__ y, int n) {
    int i = blockIdx.x * blockDim.x + threadIdx.x;
    if (i < n) y[i] = (__bf16)x[i];
}

// Pack W[K,64] (row-major f32) into this code's contiguous-k per-lane B-operand layout (same k order as the A loads) for
// v_wmma_f32_16x16x32_bf16: Wp[kt][ntile][lane][e] with
//   K = kt*32 + (lane>>4)*16 + e,  N = ntile*16 + (lane&15)
__global__ void k_pack_W(const float* __restrict__ W, __bf16* __restrict__ Wp, int K) {
    int idx = blockIdx.x * blockDim.x + threadIdx.x;
    if (idx >= K * 64) return;
    int e     = idx & 15;
    int lane  = (idx >> 4) & 31;
    int ntile = (idx >> 9) & 3;
    int kt    = idx >> 11;
    int k = kt * 32 + (lane >> 4) * 16 + e;
    int n = ntile * 16 + (lane & 15);
    Wp[idx] = (__bf16)W[k * 64 + n];
}

// ---------- WMMA GEMM: Y[N,64] = bf16(X[N,K]) @ bf16(W[K,64]) + bias ----------
// Block = 128 threads (4 waves). Wave w owns column tile w and a 64-row stripe:
// 4 independent accumulators interleave WMMAs (no D->C RAW hazard NOPs) and
// reuse each B operand 4x. Row clamping instead of exec predication.
__global__ void k_gemm_wmma(const __bf16* __restrict__ Xb, const __bf16* __restrict__ Wp,
                            const float* __restrict__ bias, float* __restrict__ Y,
                            int N, int K) {
    int lane  = threadIdx.x & 31;
    int ntile = threadIdx.x >> 5;            // 0..3
    int halfs = lane >> 4;                   // 0 | 1
    int l16   = lane & 15;
    int base  = blockIdx.x * 64;
    int r0    = base + l16;

    v8f acc[4] = {};
    int nkt = K >> 5;
    for (int kt = 0; kt < nkt; ++kt) {
        // B operand: pre-swizzled, one 32B chunk per lane; reused by 4 WMMAs.
        v16bf B = *(const v16bf*)(Wp + (((size_t)(kt * 4 + ntile) * 32 + lane) << 4));
        #pragma unroll
        for (int m = 0; m < 4; ++m) {
            int row = r0 + m * 16;
            if (row >= N) row = N - 1;       // clamp: loads valid, result unused
            // A operand (16x32 bf16): lanes<16 take K-offsets {0..15},
            // lanes>=16 take {16..31} -> two contiguous 16B chunks.
            const __bf16* p = Xb + (size_t)row * K + kt * 32 + halfs * 16;
            union AU { v16bf v; v8bf h[2]; } A;
            A.h[0] = *(const v8bf*)(p);
            A.h[1] = *(const v8bf*)(p + 8);
            acc[m] = __builtin_amdgcn_wmma_f32_16x16x32_bf16(false, A.v, false, B,
                                                             (short)0, acc[m], false, false);
        }
    }
    // C/D layout: vgpr i -> M = i + 8*halfs, N = l16
    int col = ntile * 16 + l16;
    float b = bias[col];
    #pragma unroll
    for (int m = 0; m < 4; ++m) {
        #pragma unroll
        for (int i = 0; i < 8; ++i) {
            int row = base + m * 16 + halfs * 8 + i;
            if (row < N) Y[(size_t)row * HD + col] = acc[m][i] + b;
        }
    }
}

// ---------- per-layer / per-relation init ----------
__global__ void k_init_softmax(float* __restrict__ mx, float* __restrict__ den, int n) {
    int i = blockIdx.x * blockDim.x + threadIdx.x;
    if (i < n) { mx[i] = -__builtin_inff(); den[i] = 0.f; }
}

__global__ void k_init_out(float* __restrict__ out, const float* __restrict__ bias3, int N) {
    int i = blockIdx.x * blockDim.x + threadIdx.x;
    if (i < N * HD) {
        int c = i & 63;
        out[i] = bias3[c] + bias3[64 + c] + bias3[128 + c];
    }
}

// ---------- edge pass 1: logits + segment max ----------
__global__ void k_edge_logits(const float* __restrict__ fs, const float* __restrict__ fd,
                              const float* __restrict__ attn, const int* __restrict__ src,
                              const int* __restrict__ dst, float* __restrict__ elog,
                              float* __restrict__ mx, int E) {
    int e = blockIdx.x * blockDim.x + threadIdx.x;
    if (e >= E) return;
    int s = src[e], d = dst[e];
    const float4* ps = (const float4*)(fs + (size_t)s * HD);
    const float4* pd = (const float4*)(fd + (size_t)d * HD);
    const float4* pa = (const float4*)attn;
    #pragma unroll
    for (int h = 0; h < NHEADS; ++h) {
        float lg = 0.f;
        #pragma unroll
        for (int q = 0; q < 4; ++q) {
            float4 a = ps[h * 4 + q];
            float4 b = pd[h * 4 + q];
            float4 w = pa[h * 4 + q];
            float t;
            t = a.x + b.x; lg += (t > 0.f ? t : NEG_SLOPE * t) * w.x;
            t = a.y + b.y; lg += (t > 0.f ? t : NEG_SLOPE * t) * w.y;
            t = a.z + b.z; lg += (t > 0.f ? t : NEG_SLOPE * t) * w.z;
            t = a.w + b.w; lg += (t > 0.f ? t : NEG_SLOPE * t) * w.w;
        }
        elog[(size_t)e * 4 + h] = lg;
        atomicMaxF(mx + (size_t)d * 4 + h, lg);
    }
}

// ---------- edge pass 2: exp + segment sum ----------
__global__ void k_edge_exp(const int* __restrict__ dst, float* __restrict__ elog,
                           const float* __restrict__ mx, float* __restrict__ den, int E) {
    int e = blockIdx.x * blockDim.x + threadIdx.x;
    if (e >= E) return;
    int d = dst[e];
    #pragma unroll
    for (int h = 0; h < NHEADS; ++h) {
        float ex = __expf(elog[(size_t)e * 4 + h] - mx[(size_t)d * 4 + h]);
        elog[(size_t)e * 4 + h] = ex;
        unsafeAtomicAdd(den + (size_t)d * 4 + h, ex);
    }
}

// ---------- edge pass 3: alpha-weighted scatter-add ----------
__global__ void k_edge_scatter(const float* __restrict__ fs, const int* __restrict__ src,
                               const int* __restrict__ dst, const float* __restrict__ elog,
                               const float* __restrict__ den, float* __restrict__ out, int E) {
    int e = blockIdx.x * blockDim.x + threadIdx.x;
    if (e >= E) return;
    int s = src[e], d = dst[e];
    float al[NHEADS];
    #pragma unroll
    for (int h = 0; h < NHEADS; ++h)
        al[h] = elog[(size_t)e * 4 + h] / den[(size_t)d * 4 + h];
    const float4* pf = (const float4*)(fs + (size_t)s * HD);
    float* po = out + (size_t)d * HD;
    #pragma unroll
    for (int q = 0; q < 16; ++q) {
        float4 v = pf[q];
        float a = al[q >> 2];
        unsafeAtomicAdd(po + q * 4 + 0, v.x * a);
        unsafeAtomicAdd(po + q * 4 + 1, v.y * a);
        unsafeAtomicAdd(po + q * 4 + 2, v.z * a);
        unsafeAtomicAdd(po + q * 4 + 3, v.w * a);
    }
}

// ---------- final head-mean ----------
__global__ void k_head_mean(const float* __restrict__ out, float* __restrict__ y, int N) {
    int i = blockIdx.x * blockDim.x + threadIdx.x;
    if (i < N * 16) {
        int n = i >> 4, k = i & 15;
        const float* p = out + (size_t)n * HD;
        y[i] = 0.25f * (p[k] + p[16 + k] + p[32 + k] + p[48 + k]);
    }
}

extern "C" void kernel_launch(void* const* d_in, const int* in_sizes, int n_in,
                              void* d_out, int out_size, void* d_ws, size_t ws_size,
                              hipStream_t stream) {
    const float* h    = (const float*)d_in[0];
    const int*   srcA = (const int*)d_in[1];   // [3][E]
    const int*   dstA = (const int*)d_in[2];   // [3][E]
    const int N = in_sizes[0] / 128;
    const int E = in_sizes[1] / 3;

    // Workspace carve-out (~60 MB)
    char* ws = (char*)d_ws;
    size_t off = 0;
    auto alloc = [&](size_t bytes) -> void* {
        void* p = ws + off;
        off += (bytes + 255) & ~(size_t)255;
        return p;
    };
    __bf16* xb   = (__bf16*)alloc((size_t)N * 128 * sizeof(__bf16));
    float*  fs   = (float*) alloc((size_t)N * HD * sizeof(float));
    float*  fd   = (float*) alloc((size_t)N * HD * sizeof(float));
    float*  outb = (float*) alloc((size_t)N * HD * sizeof(float));
    float*  mx   = (float*) alloc((size_t)N * 4 * sizeof(float));
    float*  den  = (float*) alloc((size_t)N * 4 * sizeof(float));
    float*  elog = (float*) alloc((size_t)E * 4 * sizeof(float));
    __bf16* wps  = (__bf16*)alloc(128 * 64 * sizeof(__bf16));
    __bf16* wpd  = (__bf16*)alloc(128 * 64 * sizeof(__bf16));

    const int T = 256;
    for (int l = 0; l < 3; ++l) {
        const int K = (l == 0) ? 128 : HD;
        const float* Wsrc = (const float*)d_in[3 + 6 * l + 0];  // [3,K,64]
        const float* Wdst = (const float*)d_in[3 + 6 * l + 1];
        const float* bsrc = (const float*)d_in[3 + 6 * l + 2];  // [3,64]
        const float* bdst = (const float*)d_in[3 + 6 * l + 3];
        const float* attn = (const float*)d_in[3 + 6 * l + 4];  // [3,4,16]
        const float* bias = (const float*)d_in[3 + 6 * l + 5];  // [3,64]

        // Activations -> bf16 (layer 0 reads h, later layers read previous out)
        const float* xsrc = (l == 0) ? h : outb;
        int nconv = N * K;
        k_f32_to_bf16<<<(nconv + T - 1) / T, T, 0, stream>>>(xsrc, xb, nconv);

        // out accumulator initialized with sum of the 3 relation biases
        k_init_out<<<(N * HD + T - 1) / T, T, 0, stream>>>(outb, bias, N);

        for (int r = 0; r < 3; ++r) {
            int npack = K * 64;
            k_pack_W<<<(npack + T - 1) / T, T, 0, stream>>>(Wsrc + (size_t)r * K * HD, wps, K);
            k_pack_W<<<(npack + T - 1) / T, T, 0, stream>>>(Wdst + (size_t)r * K * HD, wpd, K);

            int mblocks = (N + 63) / 64;
            k_gemm_wmma<<<mblocks, 128, 0, stream>>>(xb, wps, bsrc + r * HD, fs, N, K);
            k_gemm_wmma<<<mblocks, 128, 0, stream>>>(xb, wpd, bdst + r * HD, fd, N, K);

            k_init_softmax<<<(N * 4 + T - 1) / T, T, 0, stream>>>(mx, den, N * 4);

            const int* src = srcA + (size_t)r * E;
            const int* dst = dstA + (size_t)r * E;
            int eb = (E + T - 1) / T;
            k_edge_logits <<<eb, T, 0, stream>>>(fs, fd, attn + r * 64, src, dst, elog, mx, E);
            k_edge_exp    <<<eb, T, 0, stream>>>(dst, elog, mx, den, E);
            k_edge_scatter<<<eb, T, 0, stream>>>(fs, src, dst, elog, den, outb, E);
        }
    }
    k_head_mean<<<(N * 16 + T - 1) / T, T, 0, stream>>>(outb, (float*)d_out, N);
}
